// DIFFormer_25228637897372
// MI455X (gfx1250) — compile-verified, matrix operand order checked by reading, not run on hardware
//
#include <hip/hip_runtime.h>
#include <hip/hip_bf16.h>
#include <stdint.h>

// ---------------------------------------------------------------------------
// CDNA5 (gfx1250) WMMA types
// ---------------------------------------------------------------------------
typedef __attribute__((ext_vector_type(16))) __bf16 v16bf;
typedef __attribute__((ext_vector_type(8)))  float  v8f;

union BFrag { v16bf v; uint32_t u[8]; };

#define AST 136   // LDS stride (halves) for A tiles (row-major rows)
#define BST 136   // LDS stride (halves) for B tiles (column-major cols)
#define CST 40    // LDS stride (halves) for kv-kernel transposed chunks (32 rows + pad)

__device__ __forceinline__ uint16_t f32_to_bf16(float f) {
  union { float f; uint32_t u; } x; x.f = f;
  uint32_t u = x.u;
  uint32_t r = (u + 0x7FFFu + ((u >> 16) & 1u)) >> 16;  // round-to-nearest-even
  return (uint16_t)r;
}

__device__ __forceinline__ void atomAddF(float* p, float v) {
  unsafeAtomicAdd(p, v);   // hardware global_atomic_add_f32
}

// Compute NT independent 16x16 output tiles. Two tricks to avoid WMMA stalls:
//  * NT independent accumulator chains -> dependent WMMAs are NT apart.
//  * Ping-pong B-fragment buffers: loads for k-step ks+1 never write registers
//    that are sources of in-flight WMMAs (avoids the WMMA->VALU WAR hazard
//    that forced v_nop padding).
template<int NT>
__device__ __forceinline__ void gemm_tiles(const uint16_t* __restrict__ bS,
                                           const BFrag* __restrict__ afr,
                                           int ctBase, int lrow, int hi4,
                                           v8f* __restrict__ acc)
{
  const uint32_t* br[NT];
#pragma unroll
  for (int t = 0; t < NT; ++t)
    br[t] = (const uint32_t*)(bS + ((ctBase + t) * 16 + lrow) * BST);

  BFrag buf[2][NT];
#pragma unroll
  for (int t = 0; t < NT; ++t)
#pragma unroll
    for (int j = 0; j < 8; ++j)
      buf[0][t].u[j] = br[t][(j >> 2) * 8 + hi4 + (j & 3)];

#pragma unroll
  for (int t = 0; t < NT; ++t) acc[t] = (v8f){0.f,0.f,0.f,0.f,0.f,0.f,0.f,0.f};

#pragma unroll
  for (int ks = 0; ks < 4; ++ks) {
    if (ks < 3) {               // prefetch next k-step fragments into other buffer
#pragma unroll
      for (int t = 0; t < NT; ++t)
#pragma unroll
        for (int j = 0; j < 8; ++j)
          buf[(ks + 1) & 1][t].u[j] =
              br[t][(ks + 1) * 16 + (j >> 2) * 8 + hi4 + (j & 3)];
    }
#pragma unroll
    for (int t = 0; t < NT; ++t)
      acc[t] = __builtin_amdgcn_wmma_f32_16x16x32_bf16(
          false, afr[ks].v, false, buf[ks & 1][t].v, (short)0, acc[t], false, false);
  }
}

__device__ __forceinline__ void store_tile(const v8f& acc, int ct, int r0, int wave,
                                           int rAdd, int lrow,
                                           const float* __restrict__ bias,
                                           const float* __restrict__ rsv, bool useRS,
                                           float* __restrict__ outF,
                                           uint16_t* __restrict__ outBF,
                                           int M, int act, int ldOut, int ncols,
                                           bool fast)
{
  const int  col   = ct * 16 + lrow;
  const bool colOk = fast || (col < ncols);
  const float bv   = (bias != nullptr && colOk) ? bias[col] : 0.f;
#pragma unroll
  for (int r = 0; r < 8; ++r) {
    const int row = r0 + wave * 16 + rAdd + r;
    float v = acc[r];
    if (useRS) v *= rsv[r];
    v += bv;
    if      (act == 1) v = fmaxf(v, 0.f);
    else if (act == 2) v = (v > 0.f) ? (v + 1.f) : __expf(v);
    if (fast || (row < M && colOk)) {
      outF[(size_t)row * ldOut + col] = v;
      if (outBF) outBF[(size_t)row * 128 + col] = f32_to_bf16(v);
    }
  }
}

// ---------------------------------------------------------------------------
// Generic WMMA GEMM:  out[M,Ho] = act( rowScale[row] * (A[M,128] @ B[128,Ho]) + bias )
//   A bf16 row-major [M,128];  B bf16 row-major [128,Ho] (Ho multiple of 16, <=128)
//   act: 0=none, 1=relu, 2=elu+1
//   outF f32 with leading dim ldOut, valid cols = ncols; outBF optional bf16 copy (ld 128)
// ---------------------------------------------------------------------------
__global__ void __launch_bounds__(256)
wmma_gemm_kernel(const uint16_t* __restrict__ A, const uint16_t* __restrict__ B,
                 const float* __restrict__ bias, const float* __restrict__ rowScale,
                 float* __restrict__ outF, uint16_t* __restrict__ outBF,
                 int M, int Ho, int act, int ldOut, int ncols)
{
  extern __shared__ uint16_t lds[];
  uint16_t* aS = lds;                       // [128][AST]
  uint16_t* bS = lds + 128 * AST;           // [Ho][BST]  (bS[n][k])
  const int tid = threadIdx.x;
  const int r0  = blockIdx.x * 128;

  // Stage A rows r0..r0+127 (zero-pad past M): 32-bit vectorized, coalesced.
  {
    const uint32_t* Ag = (const uint32_t*)A;
    for (int i = tid; i < 128 * 64; i += 256) {
      int row = i >> 6, off = i & 63;
      uint32_t val = (r0 + row < M) ? Ag[(size_t)(r0 + row) * 64 + off] : 0u;
      *(uint32_t*)(aS + row * AST + off * 2) = val;
    }
  }
  // Stage B [128][Ho] transposed into bS[n][k].
  {
    const uint32_t* Bg = (const uint32_t*)B;
    const int Ho2 = Ho >> 1;
    for (int i = tid; i < 128 * Ho2; i += 256) {
      int k = i / Ho2, n2 = i - k * Ho2;
      uint32_t val = Bg[i];
      bS[(n2 * 2    ) * BST + k] = (uint16_t)(val & 0xFFFFu);
      bS[(n2 * 2 + 1) * BST + k] = (uint16_t)(val >> 16);
    }
  }
  __syncthreads();

  const int wave = tid >> 5;
  const int lane = tid & 31;
  const int lrow = lane & 15;
  const int hi4  = (lane >> 4) << 2;     // dword offset: lanes 16-31 hold K+8
  const int rAdd = (lane >> 4) << 3;     // D rows: lanes 16-31 -> M+8

  // A fragments for the 4 k-steps (16x32 bf16 layout, paired-K dword reads).
  BFrag afr[4];
  {
    const uint32_t* ar = (const uint32_t*)(aS + (wave * 16 + lrow) * AST);
#pragma unroll
    for (int ks = 0; ks < 4; ++ks)
#pragma unroll
      for (int j = 0; j < 8; ++j)
        afr[ks].u[j] = ar[ks * 16 + (j >> 2) * 8 + hi4 + (j & 3)];
  }

  // Hoisted per-row scale (uniform across column tiles).
  const bool useRS = (rowScale != nullptr);
  float rsv[8];
#pragma unroll
  for (int r = 0; r < 8; ++r) rsv[r] = 1.f;
  if (useRS) {
#pragma unroll
    for (int r = 0; r < 8; ++r) {
      const int row = r0 + wave * 16 + rAdd + r;
      rsv[r] = (row < M) ? rowScale[row] : 0.f;
    }
  }

  const bool fast = (r0 + 128 <= M) && (ncols == Ho);  // branch-free epilogue path
  const int  nct  = Ho >> 4;

  int ct = 0;
  for (; ct + 4 <= nct; ct += 4) {     // 4 chains: no WMMA hazard stalls
    v8f acc[4];
    gemm_tiles<4>(bS, afr, ct, lrow, hi4, acc);
#pragma unroll
    for (int t = 0; t < 4; ++t)
      store_tile(acc[t], ct + t, r0, wave, rAdd, lrow, bias, rsv, useRS,
                 outF, outBF, M, act, ldOut, ncols, fast);
  }
  if (ct + 2 <= nct) {
    v8f acc[2];
    gemm_tiles<2>(bS, afr, ct, lrow, hi4, acc);
#pragma unroll
    for (int t = 0; t < 2; ++t)
      store_tile(acc[t], ct + t, r0, wave, rAdd, lrow, bias, rsv, useRS,
                 outF, outBF, M, act, ldOut, ncols, fast);
    ct += 2;
  }
  if (ct < nct) {
    v8f acc[1];
    gemm_tiles<1>(bS, afr, ct, lrow, hi4, acc);
    store_tile(acc[0], ct, r0, wave, rAdd, lrow, bias, rsv, useRS,
               outF, outBF, M, act, ldOut, ncols, fast);
  }
}

// ---------------------------------------------------------------------------
// kv[128,128] += k[M,128]^T @ v[M,128]  (bf16 in, f32 atomic accumulate)
// 8 independent accumulator chains + ping-pong B fragments -> no WMMA stalls.
// ---------------------------------------------------------------------------
__global__ void __launch_bounds__(256)
kv_wmma_kernel(const uint16_t* __restrict__ Kb, const uint16_t* __restrict__ Vb,
               float* __restrict__ kv, int M, int rowsPerBlock)
{
  __shared__ uint16_t kT[128 * CST];   // kT[col][row-in-chunk]
  __shared__ uint16_t vT[128 * CST];
  const int tid  = threadIdx.x;
  const int wave = tid >> 5, lane = tid & 31;
  const int lrow = lane & 15;
  const int hi4  = (lane >> 4) << 2;

  v8f acc[8];
#pragma unroll
  for (int i = 0; i < 8; ++i) acc[i] = (v8f){0.f,0.f,0.f,0.f,0.f,0.f,0.f,0.f};

  const int base0 = blockIdx.x * rowsPerBlock;
  const uint32_t* Kg = (const uint32_t*)Kb;
  const uint32_t* Vg = (const uint32_t*)Vb;
  for (int c = 0; c < rowsPerBlock; c += 32) {
    const int base = base0 + c;
    if (base >= M) break;                       // uniform across the block
    for (int i = tid; i < 32 * 64; i += 256) {
      int r = i >> 6, c2 = i & 63;
      uint32_t kw = 0, vw = 0;
      if (base + r < M) {
        kw = Kg[(size_t)(base + r) * 64 + c2];
        vw = Vg[(size_t)(base + r) * 64 + c2];
      }
      kT[(c2 * 2    ) * CST + r] = (uint16_t)(kw & 0xFFFFu);
      kT[(c2 * 2 + 1) * CST + r] = (uint16_t)(kw >> 16);
      vT[(c2 * 2    ) * CST + r] = (uint16_t)(vw & 0xFFFFu);
      vT[(c2 * 2 + 1) * CST + r] = (uint16_t)(vw >> 16);
    }
    // Prefetch next chunk while this one computes (global_prefetch_b8).
    if (base + 32 + (tid >> 6) < M) {
      const size_t pi = (size_t)(base + 32 + (tid >> 6)) * 64 + (tid & 63);
      __builtin_prefetch(&Kg[pi], 0, 1);
      __builtin_prefetch(&Vg[pi], 0, 1);
    }
    __syncthreads();

    BFrag afr;                                  // A = k^T tile: A[m][kk] = kT[m][kk]
    const uint32_t* ar = (const uint32_t*)(kT + (wave * 16 + lrow) * CST);
#pragma unroll
    for (int j = 0; j < 8; ++j)
      afr.u[j] = ar[(j >> 2) * 8 + hi4 + (j & 3)];

    BFrag buf[2];
    {
      const uint32_t* br0 = (const uint32_t*)(vT + lrow * CST);
#pragma unroll
      for (int j = 0; j < 8; ++j)
        buf[0].u[j] = br0[(j >> 2) * 8 + hi4 + (j & 3)];
    }
#pragma unroll
    for (int ct = 0; ct < 8; ++ct) {
      if (ct < 7) {   // load next tile's B fragment into the other buffer
        const uint32_t* brn = (const uint32_t*)(vT + ((ct + 1) * 16 + lrow) * CST);
#pragma unroll
        for (int j = 0; j < 8; ++j)
          buf[(ct + 1) & 1].u[j] = brn[(j >> 2) * 8 + hi4 + (j & 3)];
      }
      acc[ct] = __builtin_amdgcn_wmma_f32_16x16x32_bf16(
          false, afr.v, false, buf[ct & 1].v, (short)0, acc[ct], false, false);
    }
    __syncthreads();
  }

  const int rAdd = (lane >> 4) << 3;
#pragma unroll
  for (int ct = 0; ct < 8; ++ct) {
    const int col = ct * 16 + lrow;
#pragma unroll
    for (int r = 0; r < 8; ++r)
      atomAddF(&kv[(wave * 16 + rAdd + r) * 128 + col], acc[ct][r]);
  }
}

// ---------------------------------------------------------------------------
// Small elementwise / graph kernels
// ---------------------------------------------------------------------------
__global__ void fill_f32(float* p, long long n, float v) {
  long long i = (long long)blockIdx.x * 256 + threadIdx.x;
  if (i < n) p[i] = v;
}

__global__ void convert_bf16(const float* __restrict__ in, uint16_t* __restrict__ out, long long n) {
  long long i = (long long)blockIdx.x * 256 + threadIdx.x;
  if (i < n) out[i] = f32_to_bf16(in[i]);
}

__global__ void deg_kernel(const long long* __restrict__ dst, long long E, float* __restrict__ deg) {
  long long i = (long long)blockIdx.x * 256 + threadIdx.x;
  if (i < E) atomAddF(&deg[dst[i]], 1.0f);
}

__global__ void dinv_kernel(const float* __restrict__ deg, float* __restrict__ dinv,
                            float* __restrict__ dinv2, int N) {
  int i = blockIdx.x * 256 + threadIdx.x;
  if (i < N) {
    float d = rsqrtf(fmaxf(deg[i], 1.0f));
    dinv[i] = d; dinv2[i] = d * d;
  }
}

__global__ void norm_kernel(const long long* __restrict__ src, const long long* __restrict__ dst,
                            const float* __restrict__ dinv, float* __restrict__ nrm, long long E) {
  long long i = (long long)blockIdx.x * 256 + threadIdx.x;
  if (i < E) nrm[i] = dinv[src[i]] * dinv[dst[i]];
}

__global__ void wc_pad_kernel(const float* __restrict__ Wc, uint16_t* __restrict__ out) {
  int i = blockIdx.x * 256 + threadIdx.x;          // 128 x 48 padded (valid cols 0..39)
  if (i < 128 * 48) {
    int k = i / 48, n = i - k * 48;
    out[i] = (n < 40) ? f32_to_bf16(Wc[k * 40 + n]) : (uint16_t)0;
  }
}

__global__ void bc_pad_kernel(const float* __restrict__ bc, float* __restrict__ out) {
  int i = threadIdx.x;
  if (i < 48) out[i] = (i < 40) ? bc[i] : 0.f;
}

// ksum[c] += sum over a 256-row slab of K[:,c]; column-per-thread => coalesced.
__global__ void ksum_kernel(const float* __restrict__ K, float* __restrict__ ksum, int N) {
  int c = threadIdx.x;               // 128 threads
  int r0 = blockIdx.x * 256;
  int r1 = r0 + 256; if (r1 > N) r1 = N;
  float s = 0.f;
  for (int r = r0; r < r1; ++r) {
    if (r + 32 < r1) __builtin_prefetch(&K[(size_t)(r + 32) * 128 + c], 0, 1);
    s += K[(size_t)r * 128 + c];
  }
  atomAddF(&ksum[c], s);
}

// z[i] = 1 / max(q[i,:] . ksum, 1e-6)   (one wave per row)
__global__ void zvec_kernel(const float* __restrict__ Q, const float* __restrict__ ksum,
                            float* __restrict__ z, int N) {
  int wave = threadIdx.x >> 5, lane = threadIdx.x & 31;
  int row = blockIdx.x * 8 + wave;
  if (row >= N) return;
  const float4 q4 = *(const float4*)(Q + (size_t)row * 128 + lane * 4);
  const float4 k4 = *(const float4*)(ksum + lane * 4);
  float s = q4.x * k4.x + q4.y * k4.y + q4.z * k4.z + q4.w * k4.w;
  for (int o = 16; o; o >>= 1) s += __shfl_down(s, o, 32);
  if (lane == 0) z[row] = 1.0f / fmaxf(s, 1e-6f);
}

// APPNP: z = h;  agg = dinv2 * z  (self-loop term pre-seeded)
__global__ void appnp_init(const float* __restrict__ h, const float* __restrict__ dinv2,
                           float* __restrict__ z, float* __restrict__ agg, int N) {
  long long i = (long long)blockIdx.x * 256 + threadIdx.x;   // over N*32 float4
  if (i >= (long long)N * 32) return;
  int row = (int)(i >> 5);
  float4 h4 = ((const float4*)h)[i];
  ((float4*)z)[i] = h4;
  float d2 = dinv2[row];
  float4 a4 = {d2 * h4.x, d2 * h4.y, d2 * h4.z, d2 * h4.w};
  ((float4*)agg)[i] = a4;
}

// agg[dst] += norm_e * z[src] : one wave per edge, float4 per lane (coalesced row).
__global__ void appnp_edge(const long long* __restrict__ src, const long long* __restrict__ dst,
                           const float* __restrict__ nrm, const float* __restrict__ z,
                           float* __restrict__ agg, long long E) {
  long long t = (long long)blockIdx.x * 256 + threadIdx.x;
  long long e = t >> 5;
  if (e >= E) return;
  int c4 = (int)(t & 31) * 4;
  long long s = src[e], d = dst[e];
  float w = nrm[e];
  float4 zv = *(const float4*)(z + s * 128 + c4);
  float* ap = agg + d * 128 + c4;
  atomAddF(ap + 0, w * zv.x); atomAddF(ap + 1, w * zv.y);
  atomAddF(ap + 2, w * zv.z); atomAddF(ap + 3, w * zv.w);
}

// z = 0.9*agg + 0.1*h ; agg = dinv2 * z   (re-seed self-loop for next sweep)
__global__ void appnp_update(const float* __restrict__ agg, const float* __restrict__ h,
                             const float* __restrict__ dinv2, float* __restrict__ z,
                             float* __restrict__ aggN, int N) {
  long long i = (long long)blockIdx.x * 256 + threadIdx.x;
  if (i >= (long long)N * 32) return;
  int row = (int)(i >> 5);
  float4 a4 = ((const float4*)agg)[i];
  float4 h4 = ((const float4*)h)[i];
  float4 zn = {0.9f * a4.x + 0.1f * h4.x, 0.9f * a4.y + 0.1f * h4.y,
               0.9f * a4.z + 0.1f * h4.z, 0.9f * a4.w + 0.1f * h4.w};
  ((float4*)z)[i] = zn;
  float d2 = dinv2[row];
  float4 an = {d2 * zn.x, d2 * zn.y, d2 * zn.z, d2 * zn.w};
  ((float4*)aggN)[i] = an;
}

// h = LayerNorm(h + g + d)  (one wave per row), writes f32 (in place) + bf16 copy
__global__ void ln_combine(const float* __restrict__ h, const float* __restrict__ g,
                           const float* __restrict__ d, float* __restrict__ hOut,
                           uint16_t* __restrict__ hbf, int N) {
  int wave = threadIdx.x >> 5, lane = threadIdx.x & 31;
  int row = blockIdx.x * 8 + wave;
  if (row >= N) return;
  size_t off = (size_t)row * 128 + lane * 4;
  float4 h4 = *(const float4*)(h + off);
  float4 g4 = *(const float4*)(g + off);
  float4 d4 = *(const float4*)(d + off);
  float4 s  = {h4.x + g4.x + d4.x, h4.y + g4.y + d4.y,
               h4.z + g4.z + d4.z, h4.w + g4.w + d4.w};
  float sum = s.x + s.y + s.z + s.w;
  float sq  = s.x * s.x + s.y * s.y + s.z * s.z + s.w * s.w;
  for (int o = 16; o; o >>= 1) { sum += __shfl_down(sum, o, 32); sq += __shfl_down(sq, o, 32); }
  sum = __shfl(sum, 0, 32); sq = __shfl(sq, 0, 32);
  float mean = sum * (1.f / 128.f);
  float var  = sq * (1.f / 128.f) - mean * mean;
  float inv  = rsqrtf(var + 1e-5f);
  float4 y = {(s.x - mean) * inv, (s.y - mean) * inv, (s.z - mean) * inv, (s.w - mean) * inv};
  *(float4*)(hOut + off) = y;
  uint32_t p0 = (uint32_t)f32_to_bf16(y.x) | ((uint32_t)f32_to_bf16(y.y) << 16);
  uint32_t p1 = (uint32_t)f32_to_bf16(y.z) | ((uint32_t)f32_to_bf16(y.w) << 16);
  *(uint32_t*)(hbf + off)     = p0;
  *(uint32_t*)(hbf + off + 2) = p1;
}

// ---------------------------------------------------------------------------
// Host orchestration
// ---------------------------------------------------------------------------
static inline long long cdivll(long long a, long long b) { return (a + b - 1) / b; }

extern "C" void kernel_launch(void* const* d_in, const int* in_sizes, int n_in,
                              void* d_out, int out_size, void* d_ws, size_t ws_size,
                              hipStream_t stream) {
  (void)n_in; (void)out_size; (void)ws_size;
  const float*     x    = (const float*)d_in[0];
  const long long* ei   = (const long long*)d_in[1];   // [2,E] int64
  const float*     W_in = (const float*)d_in[2];
  const float*     b_in = (const float*)d_in[3];
  const float*     Wq   = (const float*)d_in[4];
  const float*     Wk   = (const float*)d_in[5];
  const float*     Wv   = (const float*)d_in[6];
  const float*     Wo   = (const float*)d_in[7];
  const float*     bo   = (const float*)d_in[8];
  const float*     Wc   = (const float*)d_in[9];
  const float*     bc   = (const float*)d_in[10];

  const int       N = in_sizes[0] / 128;
  const long long E = (long long)in_sizes[1] / 2;
  const long long* srcI = ei;
  const long long* dstI = ei + E;
  const long long NH = (long long)N * 128;

  // ---- workspace carve (256B aligned) ----
  char* wp = (char*)d_ws;
  auto alloc = [&](size_t bytes) -> char* {
    char* p = wp; wp += (bytes + 255) & ~(size_t)255; return p;
  };
  const size_t NHf = (size_t)NH * sizeof(float);
  const size_t NHb = (size_t)NH * sizeof(uint16_t);
  float*    h    = (float*)alloc(NHf);
  float*    q    = (float*)alloc(NHf);
  float*    kF   = (float*)alloc(NHf);   // k f32, later (q@kv)*z f32
  float*    tmpF = (float*)alloc(NHf);   // v f32, later g f32
  float*    zst  = (float*)alloc(NHf);   // APPNP state -> d
  float*    agg  = (float*)alloc(NHf);
  uint16_t* hB   = (uint16_t*)alloc(NHb);
  uint16_t* qB   = (uint16_t*)alloc(NHb);
  uint16_t* kB   = (uint16_t*)alloc(NHb);  // k bf16, later t bf16
  uint16_t* tmpB = (uint16_t*)alloc(NHb);  // x bf16 -> v bf16
  float*    nrm  = (float*)alloc((size_t)E * sizeof(float));
  float*    deg  = (float*)alloc((size_t)N * sizeof(float));
  float*    dinv = (float*)alloc((size_t)N * sizeof(float));
  float*    dnv2 = (float*)alloc((size_t)N * sizeof(float));
  float*    zv   = (float*)alloc((size_t)N * sizeof(float));
  float*    kv   = (float*)alloc(128 * 128 * sizeof(float));
  float*    ksum = (float*)alloc(128 * sizeof(float));
  float*    bcp  = (float*)alloc(48 * sizeof(float));
  uint16_t* kvB  = (uint16_t*)alloc(128 * 128 * sizeof(uint16_t));
  uint16_t* WinB = (uint16_t*)alloc(128 * 128 * sizeof(uint16_t));
  uint16_t* WqB  = (uint16_t*)alloc(128 * 128 * sizeof(uint16_t));
  uint16_t* WkB  = (uint16_t*)alloc(128 * 128 * sizeof(uint16_t));
  uint16_t* WvB  = (uint16_t*)alloc(128 * 128 * sizeof(uint16_t));
  uint16_t* WoB  = (uint16_t*)alloc(128 * 128 * sizeof(uint16_t));
  uint16_t* WcB  = (uint16_t*)alloc(128 * 48 * sizeof(uint16_t));

  // ---- graph normalization (computed once) ----
  fill_f32<<<(int)cdivll(N, 256), 256, 0, stream>>>(deg, N, 1.0f);   // self-loop
  deg_kernel<<<(int)cdivll(E, 256), 256, 0, stream>>>(dstI, E, deg);
  dinv_kernel<<<(int)cdivll(N, 256), 256, 0, stream>>>(deg, dinv, dnv2, N);
  norm_kernel<<<(int)cdivll(E, 256), 256, 0, stream>>>(srcI, dstI, dinv, nrm, E);

  // ---- weights to bf16 ----
  convert_bf16<<<64, 256, 0, stream>>>(W_in, WinB, 16384);
  convert_bf16<<<64, 256, 0, stream>>>(Wq,   WqB,  16384);
  convert_bf16<<<64, 256, 0, stream>>>(Wk,   WkB,  16384);
  convert_bf16<<<64, 256, 0, stream>>>(Wv,   WvB,  16384);
  convert_bf16<<<64, 256, 0, stream>>>(Wo,   WoB,  16384);
  wc_pad_kernel<<<(128 * 48 + 255) / 256, 256, 0, stream>>>(Wc, WcB);
  bc_pad_kernel<<<1, 64, 0, stream>>>(bc, bcp);

  // ---- input projection: h = relu(x @ W_in + b_in) ----
  const int    gB     = (int)cdivll(N, 128);
  const size_t lds128 = (size_t)(128 * AST + 128 * BST) * 2;
  const size_t lds48  = (size_t)(128 * AST + 48 * BST) * 2;
  convert_bf16<<<(int)cdivll(NH, 256), 256, 0, stream>>>(x, tmpB, NH);
  wmma_gemm_kernel<<<gB, 256, lds128, stream>>>(tmpB, WinB, b_in, nullptr,
                                                h, hB, N, 128, 1, 128, 128);

  for (int layer = 0; layer < 2; ++layer) {
    // q = elu(h@Wq)+1 ; k = elu(h@Wk)+1 ; v = h@Wv
    wmma_gemm_kernel<<<gB, 256, lds128, stream>>>(hB, WqB, nullptr, nullptr,
                                                  q, qB, N, 128, 2, 128, 128);
    wmma_gemm_kernel<<<gB, 256, lds128, stream>>>(hB, WkB, nullptr, nullptr,
                                                  kF, kB, N, 128, 2, 128, 128);
    wmma_gemm_kernel<<<gB, 256, lds128, stream>>>(hB, WvB, nullptr, nullptr,
                                                  tmpF, tmpB, N, 128, 0, 128, 128);
    // kv = k^T @ v ; ksum = sum_rows(k) ; z = 1/max(q@ksum, 1e-6)
    fill_f32<<<64, 256, 0, stream>>>(kv, 128 * 128, 0.f);
    fill_f32<<<1, 128, 0, stream>>>(ksum, 128, 0.f);
    const int rpb = 416;   // 13 chunks of 32 rows per block
    kv_wmma_kernel<<<(int)cdivll(N, rpb), 256, 0, stream>>>(kB, tmpB, kv, N, rpb);
    convert_bf16<<<64, 256, 0, stream>>>(kv, kvB, 128 * 128);
    ksum_kernel<<<(int)cdivll(N, 256), 128, 0, stream>>>(kF, ksum, N);
    zvec_kernel<<<(int)cdivll(N, 8), 256, 0, stream>>>(q, ksum, zv, N);
    // t = (q @ kv) * z ;  g = t @ Wo + bo
    wmma_gemm_kernel<<<gB, 256, lds128, stream>>>(qB, kvB, nullptr, zv,
                                                  kF, kB, N, 128, 0, 128, 128);
    wmma_gemm_kernel<<<gB, 256, lds128, stream>>>(kB, WoB, bo, nullptr,
                                                  tmpF, nullptr, N, 128, 0, 128, 128);
    // APPNP: 10 propagation sweeps
    appnp_init<<<(int)cdivll(NH / 4, 256), 256, 0, stream>>>(h, dnv2, zst, agg, N);
    for (int it = 0; it < 10; ++it) {
      appnp_edge<<<(int)cdivll(E * 32, 256), 256, 0, stream>>>(srcI, dstI, nrm, zst, agg, E);
      appnp_update<<<(int)cdivll(NH / 4, 256), 256, 0, stream>>>(agg, h, dnv2, zst, agg, N);
    }
    // h = LayerNorm(h + g + d)
    ln_combine<<<(int)cdivll(N, 8), 256, 0, stream>>>(h, tmpF, zst, h, hB, N);
  }

  // ---- classifier: out = h @ Wc + bc  (padded to 48 cols, 40 valid) ----
  wmma_gemm_kernel<<<gB, 256, lds48, stream>>>(hB, WcB, bcp, nullptr,
                                               (float*)d_out, nullptr, N, 48, 0, 40, 40);
}
